// AttentionHead_4964982194379
// MI455X (gfx1250) — compile-verified, matrix operand order checked by reading, not run on hardware
//
#include <hip/hip_runtime.h>

typedef __attribute__((ext_vector_type(16))) _Float16 v16h;
typedef __attribute__((ext_vector_type(8)))  _Float16 v8h;
typedef __attribute__((ext_vector_type(8)))  float    v8f;

#define DIM_IN 768
#define DIM_H  64
#define SEQ    2048
#define NBATCH 8
#define BK     32      // keys per attention inner block

#if defined(__gfx1250__)
#define ASYNC_STAGE 1
#else
#define ASYNC_STAGE 0
#endif

__device__ __forceinline__ v8f wmma_f16(v16h a, v16h b, v8f c) {
  // D = A(16x32 f16) * B(32x16 f16) + C(16x16 f32)
  return __builtin_amdgcn_wmma_f32_16x16x32_f16(false, a, false, b, (short)0, c,
                                                false, false);
}

// 16-byte global -> LDS copy. Async data-mover path on gfx1250 (ASYNCcnt),
// VGPR staging fallback otherwise.
__device__ __forceinline__ void stage16(const _Float16* gsrc, _Float16* ldst) {
#if ASYNC_STAGE
  unsigned lds = (unsigned)(uintptr_t)ldst;   // LDS aperture: addr[31:0] = LDS offset
  asm volatile("global_load_async_to_lds_b128 %0, %1, off"
               :: "v"(lds), "v"(gsrc) : "memory");
#else
  *(v8h*)ldst = *(const v8h*)gsrc;
#endif
}

__device__ __forceinline__ void stage_wait() {
#if ASYNC_STAGE
  asm volatile("s_wait_asynccnt 0x0" ::: "memory");
#endif
}

// A-tile (16x32) from row-major f32 [M x K], base at (row0,k0), row stride ld.
__device__ __forceinline__ v16h load_a_f32(const float* base, int ld) {
  int lane = threadIdx.x & 31;
  const float* p = base + (lane & 15) * ld + ((lane & 16) ? 8 : 0);
  v16h r;
#pragma unroll
  for (int i = 0; i < 8; ++i) r[i] = (_Float16)p[i];
#pragma unroll
  for (int i = 0; i < 8; ++i) r[8 + i] = (_Float16)p[16 + i];
  return r;
}

// A-tile (or B-tile from an N-major source) from row-major f16, row stride ld.
__device__ __forceinline__ v16h load_t_f16(const _Float16* base, int ld) {
  int lane = threadIdx.x & 31;
  const _Float16* p = base + (lane & 15) * ld + ((lane & 16) ? 8 : 0);
  v16h r;
#pragma unroll
  for (int i = 0; i < 8; ++i) r[i] = p[i];
#pragma unroll
  for (int i = 0; i < 8; ++i) r[8 + i] = p[16 + i];
  return r;
}

// --- one-shot weight pack: f32 [768x64] row-major -> f16 in exact WMMA B-tile
// lane order: [k-step(24)][n-tile(4)][lane(32)][half(16)]  (each tile = 32B/lane)
__global__ __launch_bounds__(256)
void pack_w_kernel(const float* __restrict__ W, _Float16* __restrict__ out) {
  int idx  = blockIdx.x * 256 + threadIdx.x;     // 0 .. 768*64-1
  int i    = idx & 15;
  int lane = (idx >> 4) & 31;
  int tile = (idx >> 9) & 3;
  int k0   = idx >> 11;                          // 0..23
  int kb   = (lane & 16) ? 8 : 0;
  int kk   = (i < 8) ? (kb + i) : (kb + 8 + i);  // i>=8 -> kb+16+(i-8)
  int k    = k0 * 32 + kk;
  int n    = tile * 16 + (lane & 15);
  out[idx] = (_Float16)W[k * DIM_H + n];
}

// mode 0: plain f16 row-major; 1: L2-normalize each row then f16 row-major;
// 2: store transposed per batch as [B][64][S] f16.
__global__ __launch_bounds__(32)
void proj_kernel(const float* __restrict__ X, const _Float16* __restrict__ pW,
                 const float* __restrict__ bias, _Float16* __restrict__ out,
                 int mode) {
  __shared__ float tileBuf[16 * DIM_H];
  int lane = threadIdx.x & 31;
  int row0 = blockIdx.x * 16;

  const v16h* Wp = (const v16h*)pW;              // 32B-aligned packed tiles

  v8f acc[4] = {};
  for (int k0 = 0; k0 < DIM_IN; k0 += 32) {
    if (k0 + 64 < DIM_IN)
      __builtin_prefetch(X + row0 * DIM_IN + k0 + 64, 0, 1);
    v16h a = load_a_f32(X + row0 * DIM_IN + k0, DIM_IN);
    int tbase = (k0 >> 5) << 2;
#pragma unroll
    for (int t = 0; t < 4; ++t) {
      v16h b = Wp[(tbase + t) * 32 + lane];
      acc[t] = wmma_f16(a, b, acc[t]);
    }
  }

  int col   = lane & 15;
  int rbase = (lane & 16) ? 8 : 0;
#pragma unroll
  for (int t = 0; t < 4; ++t) {
    float bv = bias[t * 16 + col];
#pragma unroll
    for (int r = 0; r < 8; ++r)
      tileBuf[(rbase + r) * DIM_H + t * 16 + col] = acc[t][r] + bv;
  }
  __syncthreads();

  if (mode == 1) {
    if (lane < 16) {
      float ss = 0.f;
      for (int c = 0; c < DIM_H; ++c) {
        float x = tileBuf[lane * DIM_H + c];
        ss += x * x;
      }
      float inv = rsqrtf(ss);
      for (int c = 0; c < DIM_H; ++c)
        out[(row0 + lane) * DIM_H + c] = (_Float16)(tileBuf[lane * DIM_H + c] * inv);
    }
  } else if (mode == 2) {
    int b  = row0 >> 11;                         // tiles never cross batches
    int s0 = row0 & (SEQ - 1);
    for (int i = lane; i < 16 * DIM_H; i += 32) {
      int r = i >> 6, c = i & 63;
      out[((size_t)b * DIM_H + c) * SEQ + s0 + r] = (_Float16)tileBuf[r * DIM_H + c];
    }
  } else {
    for (int i = lane; i < 16 * DIM_H; i += 32) {
      int r = i >> 6, c = i & 63;
      out[(row0 + r) * DIM_H + c] = (_Float16)tileBuf[r * DIM_H + c];
    }
  }
}

// 8 waves per block; each wave owns one 16-row query tile; K/V staged in LDS
// (double-buffered, async data-mover) and shared by all 8 waves.
__global__ __launch_bounds__(256)
void attn_kernel(const _Float16* __restrict__ qn, const _Float16* __restrict__ kn,
                 const _Float16* __restrict__ vT, float* __restrict__ out) {
  __shared__ _Float16 kTile[2][BK * DIM_H];      // [key][feat]   4KB each
  __shared__ _Float16 vTile[2][DIM_H * BK];      // [outcol][key] 4KB each
  __shared__ _Float16 pbuf[8][16 * 32];          // per-wave P transpose, 8KB

  int tid  = threadIdx.x;
  int lane = tid & 31;
  int wave = tid >> 5;
  int b    = blockIdx.y;
  int row0 = blockIdx.x * 128 + wave * 16;

  const _Float16* qb  = qn + (size_t)b * SEQ * DIM_H;
  const _Float16* kbp = kn + (size_t)b * SEQ * DIM_H;
  const _Float16* vbp = vT + (size_t)b * DIM_H * SEQ;

  v16h aq0 = load_t_f16(qb + row0 * DIM_H, DIM_H);        // K 0..31
  v16h aq1 = load_t_f16(qb + row0 * DIM_H + 32, DIM_H);   // K 32..63

  int vr = tid >> 2, vc = (tid & 3) * 8;        // staging map for vTile
  // stage j=0 into buffer 0 (kn rows contiguous: one flat 4KB copy)
  stage16(kbp + tid * 8, &kTile[0][tid * 8]);
  stage16(vbp + (size_t)vr * SEQ + vc, &vTile[0][vr * BK + vc]);

  v8f o[4] = {};
  float m[8], lsum[8];
#pragma unroll
  for (int r = 0; r < 8; ++r) { m[r] = -3.0e38f; lsum[r] = 0.f; }

  int col   = lane & 15;
  int rbase = (lane & 16) ? 8 : 0;
  const float scale = 0.125f;                    // 1/sqrt(64)

  int buf = 0;
  for (int j = 0; j < SEQ; j += BK, buf ^= 1) {
    stage_wait();                                // my async copies into `buf` landed
    __syncthreads();                             // everyone's copies landed
    if (j + BK < SEQ) {                          // stage next block into buf^1
      int jn = j + BK;
      stage16(kbp + jn * 64 + tid * 8, &kTile[buf ^ 1][tid * 8]);
      stage16(vbp + (size_t)vr * SEQ + jn + vc, &vTile[buf ^ 1][vr * BK + vc]);
      if (jn + BK < SEQ) {
        __builtin_prefetch(kbp + (jn + BK) * 64 + tid * 8, 0, 1);
        __builtin_prefetch(vbp + (size_t)vr * SEQ + jn + BK + vc, 0, 1);
      }
    }

    const _Float16* kT  = kTile[buf];
    const _Float16* vTl = vTile[buf];
    _Float16*       pb  = pbuf[wave];

    // scores tile: [16 query rows x 32 keys], two 16x16 f32 accumulators
    v8f s0 = {}, s1 = {};
    s0 = wmma_f16(aq0, load_t_f16(kT, DIM_H), s0);
    s0 = wmma_f16(aq1, load_t_f16(kT + 32, DIM_H), s0);
    s1 = wmma_f16(aq0, load_t_f16(kT + 16 * DIM_H, DIM_H), s1);
    s1 = wmma_f16(aq1, load_t_f16(kT + 16 * DIM_H + 32, DIM_H), s1);

#pragma unroll
    for (int r = 0; r < 8; ++r) {
      float a0 = s0[r] * scale;
      float a1 = s1[r] * scale;
      float smax = fmaxf(a0, a1);
      smax = fmaxf(smax, __shfl_xor(smax, 1, 32));
      smax = fmaxf(smax, __shfl_xor(smax, 2, 32));
      smax = fmaxf(smax, __shfl_xor(smax, 4, 32));
      smax = fmaxf(smax, __shfl_xor(smax, 8, 32));  // max over 16-lane half
      float mn    = fmaxf(m[r], smax);
      float alpha = __expf(m[r] - mn);              // 0 on first iteration
      m[r] = mn;
      float p0 = __expf(a0 - mn);
      float p1 = __expf(a1 - mn);
      lsum[r] = lsum[r] * alpha + p0 + p1;          // lane-local partial sum
      o[0][r] *= alpha; o[1][r] *= alpha; o[2][r] *= alpha; o[3][r] *= alpha;
      pb[(rbase + r) * 32 + col]      = (_Float16)p0;
      pb[(rbase + r) * 32 + 16 + col] = (_Float16)p1;
    }

    // reload P in A-matrix layout (intra-wave LDS ordering is in-order)
    v16h ap;
    {
      int rowp = lane & 15;
      int kbo  = (lane & 16) ? 8 : 0;
#pragma unroll
      for (int i = 0; i < 8; ++i) ap[i] = pb[rowp * 32 + kbo + i];
#pragma unroll
      for (int i = 0; i < 8; ++i) ap[8 + i] = pb[rowp * 32 + kbo + 16 + i];
    }

    // O += P[16x32] * V[32x64]  (vTile rows contiguous in key index)
#pragma unroll
    for (int t = 0; t < 4; ++t)
      o[t] = wmma_f16(ap, load_t_f16(vTl + (t * 16) * BK, BK), o[t]);
  }

  // reduce row sums across the 16-lane halves
#pragma unroll
  for (int r = 0; r < 8; ++r) {
    float s = lsum[r];
    s += __shfl_xor(s, 1, 32);
    s += __shfl_xor(s, 2, 32);
    s += __shfl_xor(s, 4, 32);
    s += __shfl_xor(s, 8, 32);
    lsum[r] = s;
  }

  float* ob = out + ((size_t)b * SEQ + row0) * DIM_H;
#pragma unroll
  for (int t = 0; t < 4; ++t)
#pragma unroll
    for (int r = 0; r < 8; ++r)
      ob[(rbase + r) * DIM_H + t * 16 + col] = o[t][r] / lsum[r];
}

extern "C" void kernel_launch(void* const* d_in, const int* in_sizes, int n_in,
                              void* d_out, int out_size, void* d_ws, size_t ws_size,
                              hipStream_t stream) {
  const float* query = (const float*)d_in[0];
  const float* key   = (const float*)d_in[1];
  const float* value = (const float*)d_in[2];
  const float* Wq    = (const float*)d_in[3];
  const float* bq    = (const float*)d_in[4];
  const float* Wk    = (const float*)d_in[5];
  const float* bk    = (const float*)d_in[6];
  const float* Wv    = (const float*)d_in[7];
  const float* bv    = (const float*)d_in[8];
  float* out = (float*)d_out;

  const size_t n  = (size_t)NBATCH * SEQ * DIM_H;   // 1M f16 elements each
  const size_t wn = (size_t)DIM_IN * DIM_H;         // 48K f16 elements each
  _Float16* qn  = (_Float16*)d_ws;
  _Float16* kn  = qn + n;
  _Float16* vT  = kn + n;
  _Float16* pWq = vT + n;
  _Float16* pWk = pWq + wn;
  _Float16* pWv = pWk + wn;

  // pack weights once into WMMA B-tile order (f16)
  int packBlocks = (int)(wn / 256);                 // 192
  pack_w_kernel<<<packBlocks, 256, 0, stream>>>(Wq, pWq);
  pack_w_kernel<<<packBlocks, 256, 0, stream>>>(Wk, pWk);
  pack_w_kernel<<<packBlocks, 256, 0, stream>>>(Wv, pWv);

  dim3 blk32(32);
  int rowTiles = NBATCH * SEQ / 16;                 // 1024
  proj_kernel<<<rowTiles, blk32, 0, stream>>>(query, pWq, bq, qn, 1);
  proj_kernel<<<rowTiles, blk32, 0, stream>>>(key,   pWk, bk, kn, 1);
  proj_kernel<<<rowTiles, blk32, 0, stream>>>(value, pWv, bv, vT, 2);

  attn_kernel<<<dim3(SEQ / 128, NBATCH), 256, 0, stream>>>(qn, kn, vT, out);
}
